// SequenceQuantizer_89867895701684
// MI455X (gfx1250) — matrix operand order repaired and checked by reading, not compiled, not run on hardware
//
#include <hip/hip_runtime.h>
#include <math.h>

// ---------------------------------------------------------------------------
// VQ sequence quantizer for MI455X (gfx1250), wave32 + WMMA bf16 + async LDS.
// N = 8192 tokens, D = 1024, K = 8192 codes.
// ---------------------------------------------------------------------------

#define N_TOK 8192
#define DIM   1024
#define K_CODE 8192

typedef __bf16 v16bf __attribute__((ext_vector_type(16)));
typedef __bf16 v4bf  __attribute__((ext_vector_type(4)));
typedef float  v8f   __attribute__((ext_vector_type(8)));

union Frag16 {
    v16bf v;
    struct { uint4 lo; uint4 hi; } u;
};

// ---- deterministic block reduction (256 threads) ---------------------------
__device__ __forceinline__ float blockReduceSum(float v) {
    __shared__ float red[8];
    __syncthreads();                       // protect shared across repeated calls
    const int lane = threadIdx.x & 31;
    const int w    = threadIdx.x >> 5;
    #pragma unroll
    for (int o = 16; o > 0; o >>= 1) v += __shfl_xor(v, o, 32);
    if (lane == 0) red[w] = v;
    __syncthreads();
    v = (threadIdx.x < 8) ? red[threadIdx.x] : 0.0f;
    if (w == 0) {
        #pragma unroll
        for (int o = 4; o > 0; o >>= 1) v += __shfl_xor(v, o, 32);
    }
    return v;  // valid on thread 0
}

// ---- kernel 0: zero counts -------------------------------------------------
__global__ __launch_bounds__(256) void vq_zero(unsigned int* __restrict__ counts) {
    int i = blockIdx.x * 256 + threadIdx.x;
    if (i < K_CODE) counts[i] = 0u;
}

// ---- kernel 1: f32 -> bf16 convert + row squared-norms ---------------------
// grid = 16384 blocks (rows 0..8191 = inputs, 8192..16383 = codebook)
__global__ __launch_bounds__(256) void vq_prep(const float* __restrict__ x,
                                               const float* __restrict__ cbk,
                                               __bf16* __restrict__ xb,
                                               __bf16* __restrict__ cbb,
                                               float* __restrict__ xnorm,
                                               float* __restrict__ cnorm) {
    const int row = blockIdx.x;
    const int tid = threadIdx.x;
    const float* src;
    __bf16*      dst;
    float*       nrm;
    int          r;
    if (row < N_TOK) {
        r = row; src = x + (size_t)r * DIM; dst = xb + (size_t)r * DIM; nrm = xnorm;
    } else {
        r = row - N_TOK; src = cbk + (size_t)r * DIM; dst = cbb + (size_t)r * DIM; nrm = cnorm;
    }
    float4 v = ((const float4*)src)[tid];          // 4 dims per thread
    float s = v.x * v.x + v.y * v.y + v.z * v.z + v.w * v.w;
    v4bf o;
    o[0] = (__bf16)v.x; o[1] = (__bf16)v.y; o[2] = (__bf16)v.z; o[3] = (__bf16)v.w;
    ((v4bf*)dst)[tid] = o;
    s = blockReduceSum(s);
    if (tid == 0) nrm[r] = s;
}

// ---- kernel 2: WMMA distance + packed argmin -------------------------------
// grid = 128 blocks, 256 threads (8 waves). Block owns 64 tokens (A in LDS).
// Wave w sweeps 2 code tiles per iter: codes j*256 + w*32 + {0..31}, j in [0,32).
#define LDS_ROW_B 2064                     // 2048 B data + 16 B pad: stride = 516 dw = 4 banks
#define LDS_A_BYTES (64 * LDS_ROW_B)       // 132096
#define LDS_TOTAL_BYTES (LDS_A_BYTES + 64 * 4)

__global__ __launch_bounds__(256, 1) void vq_argmin(const __bf16* __restrict__ xb,
                                                    const __bf16* __restrict__ cbb,
                                                    const float* __restrict__ xnorm,
                                                    const float* __restrict__ cnorm,
                                                    unsigned int* __restrict__ bestIdx) {
    extern __shared__ char smem[];
    char*         ldsA    = smem;
    unsigned int* ldsBest = (unsigned int*)(smem + LDS_A_BYTES);

    const int tid    = threadIdx.x;
    const int wave   = tid >> 5;
    const int lane   = tid & 31;
    const int lane16 = lane & 15;
    const int hlf    = lane >> 4;
    const int mBase  = blockIdx.x * 64;

    if (tid < 64) ldsBest[tid] = 0xFFFFFFFFu;

    // --- async-stage 64 token rows (bf16) into LDS, padded row stride -------
    // Each thread copies 32 x 16 B; ASYNCcnt tracks completion (CDNA5 path).
    {
        const int row = tid >> 2;          // 0..63
        const int seg = tid & 3;           // quarter of a 2048-B row
        unsigned ldsAddr = (unsigned)(uintptr_t)(void*)ldsA + row * LDS_ROW_B + seg * 512;
        unsigned long long g = (unsigned long long)(uintptr_t)xb
                             + (unsigned long long)(mBase + row) * 2048ull + (unsigned)(seg * 512);
        #pragma unroll 4
        for (int i = 0; i < 32; ++i) {
            asm volatile("global_load_async_to_lds_b128 %0, %1, off"
                         :: "v"(ldsAddr), "v"(g) : "memory");
            ldsAddr += 16;
            g += 16;
        }
        asm volatile("s_wait_asynccnt 0x0" ::: "memory");
    }
    __syncthreads();

    // per-lane x-norms for the 4 row-tiles (C-layout: VGPR r -> M = r + 8*hlf)
    float xn[4][8];
    #pragma unroll
    for (int rt = 0; rt < 4; ++rt)
        #pragma unroll
        for (int r = 0; r < 8; ++r)
            xn[rt][r] = xnorm[mBase + rt * 16 + hlf * 8 + r];

    unsigned int runmin[4][8];
    #pragma unroll
    for (int rt = 0; rt < 4; ++rt)
        #pragma unroll
        for (int r = 0; r < 8; ++r) runmin[rt][r] = 0xFFFFFFFFu;

    // A-fragment base byte offsets (invariant over j): rows rt*16+lane16
    const char* aBase[4];
    #pragma unroll
    for (int rt = 0; rt < 4; ++rt)
        aBase[rt] = ldsA + (rt * 16 + lane16) * LDS_ROW_B + hlf * 32;

    #pragma unroll 1
    for (int j = 0; j < 32; ++j) {
        const int code0 = j * 256 + wave * 32 + lane16;   // two B columns per lane
        const int code1 = code0 + 16;
        const float cn0 = cnorm[code0];
        const float cn1 = cnorm[code1];
        const __bf16* b0 = cbb + (size_t)code0 * DIM + hlf * 16;
        const __bf16* b1 = cbb + (size_t)code1 * DIM + hlf * 16;

        v8f acc[4][2];
        #pragma unroll
        for (int rt = 0; rt < 4; ++rt) {
            acc[rt][0] = (v8f)(0.0f);
            acc[rt][1] = (v8f)(0.0f);
        }

        #pragma unroll 2
        for (int dc = 0; dc < 32; ++dc) {
            Frag16 fb0, fb1;                              // B 32x16 bf16 fragments
            fb0.u.lo = *(const uint4*)(b0 + dc * 32);
            fb0.u.hi = *(const uint4*)(b0 + dc * 32 + 8);
            fb1.u.lo = *(const uint4*)(b1 + dc * 32);
            fb1.u.hi = *(const uint4*)(b1 + dc * 32 + 8);

            Frag16 fa[4];                                 // A 16x32 bf16 fragments (LDS)
            #pragma unroll
            for (int rt = 0; rt < 4; ++rt) {
                const char* ap = aBase[rt] + dc * 64;
                fa[rt].u.lo = *(const uint4*)ap;
                fa[rt].u.hi = *(const uint4*)(ap + 16);
            }
            #pragma unroll
            for (int rt = 0; rt < 4; ++rt) {
                acc[rt][0] = __builtin_amdgcn_wmma_f32_16x16x32_bf16(
                    false, fa[rt].v, false, fb0.v, (short)0, acc[rt][0], false, false);
                acc[rt][1] = __builtin_amdgcn_wmma_f32_16x16x32_bf16(
                    false, fa[rt].v, false, fb1.v, (short)0, acc[rt][1], false, false);
            }
        }

        // epilogue: dist = ||x||^2 + ||c||^2 - 2 x.c ; pack (dist | idx) ; run min
        const unsigned int cb0 = (unsigned int)code0;
        const unsigned int cb1 = (unsigned int)code1;
        #pragma unroll
        for (int rt = 0; rt < 4; ++rt)
            #pragma unroll
            for (int r = 0; r < 8; ++r) {
                float d0 = fmaxf(fmaf(-2.0f, acc[rt][0][r], xn[rt][r] + cn0), 0.0f);
                float d1 = fmaxf(fmaf(-2.0f, acc[rt][1][r], xn[rt][r] + cn1), 0.0f);
                unsigned int p0 = (__float_as_uint(d0) & 0xFFFFE000u) | cb0;
                unsigned int p1 = (__float_as_uint(d1) & 0xFFFFE000u) | cb1;
                if (code0 == 0) p0 = 0xFFFFFFFFu;         // code 0 forbidden for non-pad
                unsigned int m = p0 < p1 ? p0 : p1;
                runmin[rt][r] = m < runmin[rt][r] ? m : runmin[rt][r];
            }
    }

    // min across the 16 lanes of each half (N dimension), then across waves
    #pragma unroll
    for (int rt = 0; rt < 4; ++rt)
        #pragma unroll
        for (int r = 0; r < 8; ++r) {
            unsigned int v = runmin[rt][r];
            #pragma unroll
            for (int s = 1; s < 16; s <<= 1) {
                unsigned int o = (unsigned int)__shfl_xor((int)v, s, 32);
                v = o < v ? o : v;
            }
            runmin[rt][r] = v;
        }
    if (lane16 == 0) {
        #pragma unroll
        for (int rt = 0; rt < 4; ++rt)
            #pragma unroll
            for (int r = 0; r < 8; ++r)
                atomicMin(&ldsBest[rt * 16 + hlf * 8 + r], runmin[rt][r]);
    }
    __syncthreads();
    if (tid < 64) bestIdx[mBase + tid] = ldsBest[tid] & 8191u;
}

// ---- kernel 3: gather quantized, per-token sq-err, usage counts ------------
// grid = 8192 blocks (one per token), 256 threads (4 dims each)
__global__ __launch_bounds__(256) void vq_gather(const float* __restrict__ x,
                                                 const unsigned char* __restrict__ mask,
                                                 const float* __restrict__ cbk,
                                                 const unsigned int* __restrict__ bestIdx,
                                                 float* __restrict__ outQ,
                                                 float* __restrict__ outIdx,
                                                 unsigned int* __restrict__ counts,
                                                 float* __restrict__ tokenSums) {
    const int t   = blockIdx.x;
    const int tid = threadIdx.x;
    const bool pad = mask[t] != 0;
    const unsigned int idx = pad ? 0u : bestIdx[t];

    float4 q  = ((const float4*)(cbk + (size_t)idx * DIM))[tid];
    float4 xv = ((const float4*)(x + (size_t)t * DIM))[tid];
    ((float4*)(outQ + (size_t)t * DIM))[tid] = q;        // quantized_st == quantized (values)

    float dx = q.x - xv.x, dy = q.y - xv.y, dz = q.z - xv.z, dw = q.w - xv.w;
    float s = dx * dx + dy * dy + dz * dz + dw * dw;
    s = blockReduceSum(s);
    if (tid == 0) {
        tokenSums[t] = s;
        outIdx[t] = (float)idx;
        atomicAdd(&counts[idx], 1u);
    }
}

// ---- kernel 4: loss + perplexity (single block, deterministic) -------------
__global__ __launch_bounds__(256) void vq_finalize(const unsigned int* __restrict__ counts,
                                                   const float* __restrict__ tokenSums,
                                                   const unsigned char* __restrict__ mask,
                                                   float* __restrict__ outLoss,
                                                   float* __restrict__ outPerp) {
    const int tid = threadIdx.x;
    float ssum = 0.0f, np = 0.0f, H = 0.0f;
    for (int i = tid; i < N_TOK; i += 256) {
        ssum += tokenSums[i];
        np   += (mask[i] != 0) ? 0.0f : 1.0f;
        float p = (float)counts[i] * (1.0f / (float)N_TOK);
        H += p * logf(p + 1e-10f);
    }
    // note: K_CODE == N_TOK == 8192, single strided loop covers both
    ssum = blockReduceSum(ssum);
    np   = blockReduceSum(np);
    H    = blockReduceSum(H);
    if (tid == 0) {
        *outLoss = 1.25f * ssum / (np * (float)DIM);     // q_latent + 0.25*e_latent
        *outPerp = expf(-H);
    }
}

// ---------------------------------------------------------------------------
extern "C" void kernel_launch(void* const* d_in, const int* in_sizes, int n_in,
                              void* d_out, int out_size, void* d_ws, size_t ws_size,
                              hipStream_t stream) {
    const float*         x    = (const float*)d_in[0];          // [4,2048,1024]
    const unsigned char* mask = (const unsigned char*)d_in[1];  // [4,2048] bool
    const float*         cbk  = (const float*)d_in[2];          // [8192,1024]
    float* out = (float*)d_out;

    char* ws = (char*)d_ws;
    __bf16*       xb        = (__bf16*)ws;                              // 16 MiB
    __bf16*       cbb       = (__bf16*)(ws + 16777216);                 // 16 MiB
    float*        xnorm     = (float*)(ws + 33554432);                  // 32 KiB
    float*        cnorm     = (float*)(ws + 33587200);                  // 32 KiB
    unsigned int* bestIdx   = (unsigned int*)(ws + 33619968);           // 32 KiB
    unsigned int* counts    = (unsigned int*)(ws + 33652736);           // 32 KiB
    float*        tokenSums = (float*)(ws + 33685504);                  // 32 KiB

    float* outQ    = out;                       // 8388608 floats
    float* outIdx  = out + 8388608;             // 8192 floats
    float* outLoss = out + 8396800;             // 1
    float* outPerp = out + 8396801;             // 1

    vq_zero<<<dim3(32), dim3(256), 0, stream>>>(counts);
    vq_prep<<<dim3(16384), dim3(256), 0, stream>>>(x, cbk, xb, cbb, xnorm, cnorm);
    vq_argmin<<<dim3(128), dim3(256), LDS_TOTAL_BYTES, stream>>>(xb, cbb, xnorm, cnorm, bestIdx);
    vq_gather<<<dim3(8192), dim3(256), 0, stream>>>(x, mask, cbk, bestIdx,
                                                    outQ, outIdx, counts, tokenSums);
    vq_finalize<<<dim3(1), dim3(256), 0, stream>>>(counts, tokenSums, mask, outLoss, outPerp);
}
